// Pew_LSTM_layer_longmem_8950711845908
// MI455X (gfx1250) — compile-verified, matrix operand order checked by reading, not run on hardware
//
#include <hip/hip_runtime.h>
#include <hip/hip_bf16.h>
#include <math.h>

typedef __attribute__((ext_vector_type(16))) _Float16 v16h;
typedef __attribute__((ext_vector_type(8)))  _Float16 v8h;
typedef __attribute__((ext_vector_type(8)))  float    v8f;
typedef __attribute__((ext_vector_type(4)))  int      v4i;

constexpr int kD = 90, kT = 96, kI = 512, kH = 1024, kW = 64;
constexpr int kN = kD * kT;   // 8640

#if __has_builtin(__builtin_amdgcn_global_load_async_to_lds_b128) && \
    __has_builtin(__builtin_amdgcn_s_wait_asynccnt)
#define USE_ASYNC_LDS 1
#else
#define USE_ASYNC_LDS 0
#endif

// LDS tile geometry: 16 rows x 32 halfs, row stride padded to 80B (bank spread)
constexpr int kTileRowB = 80;
constexpr int kTileB    = 16 * kTileRowB;   // 1280 bytes per staged tile

// ---------------------------------------------------------------- helpers

__device__ __forceinline__ float fast_rcp(float x) {
#if __has_builtin(__builtin_amdgcn_rcpf)
  return __builtin_amdgcn_rcpf(x);
#else
  return 1.0f / x;
#endif
}
__device__ __forceinline__ float sigmoid_fast(float x) {
  return fast_rcp(1.0f + __expf(-x));         // v_exp_f32 + v_rcp_f32
}
__device__ __forceinline__ float tanh_fast(float x) {
  return 1.0f - 2.0f * fast_rcp(__expf(2.0f * x) + 1.0f);
}

// Direct-from-global 16x32 f16 fragment (A-layout; also B via mirrored lane->N).
//   lanes 0-15 : row = lane,    K in {k0+0..7, k0+16..23}
//   lanes 16-31: row = lane-16, K in {k0+8..15, k0+24..31}
__device__ __forceinline__ v16h load_frag(const _Float16* __restrict__ base,
                                          int ld, int row0, int k0, int lane) {
  const int r  = lane & 15;
  const int kc = k0 + ((lane >> 4) << 3);
  const _Float16* p = base + (size_t)(row0 + r) * (size_t)ld + kc;
  v8h lo = *(const v8h*)(p);
  v8h hi = *(const v8h*)(p + 16);
  return __builtin_shufflevector(lo, hi, 0,1,2,3,4,5,6,7,8,9,10,11,12,13,14,15);
}

// Fragment from a staged LDS tile (row-major, kTileRowB stride; 16B chunks are
// K-cols {0-7, 8-15, 16-23, 24-31}).
__device__ __forceinline__ v16h frag_from_lds(const char* lds, int tile, int lane) {
  const int r  = lane & 15;
  const int kc = (lane >> 4) << 4;            // 0 or 16 bytes
  const char* p = lds + tile * kTileB + r * kTileRowB + kc;
  v8h lo = *(const v8h*)(p);                  // K chunk 0/1
  v8h hi = *(const v8h*)(p + 32);             // K chunk 2/3
  return __builtin_shufflevector(lo, hi, 0,1,2,3,4,5,6,7,8,9,10,11,12,13,14,15);
}

__device__ __forceinline__ v8f wmma_f16(v16h a, v16h b, v8f c) {
  return __builtin_amdgcn_wmma_f32_16x16x32_f16(false, a, false, b,
                                                (short)0, c, false, false);
}

__device__ __forceinline__ void stage_chunk(const _Float16* g, char* ldst) {
#if USE_ASYNC_LDS
  __builtin_amdgcn_global_load_async_to_lds_b128(
      (__attribute__((address_space(1))) v4i*)g,
      (__attribute__((address_space(3))) v4i*)ldst, 0, 0);
#else
  *(v8h*)ldst = *(const v8h*)g;
#endif
}

__device__ __forceinline__ const _Float16* pick7(
    const _Float16* p0, const _Float16* p1, const _Float16* p2,
    const _Float16* p3, const _Float16* p4, const _Float16* p5,
    const _Float16* p6, int t) {
  switch (t) {
    case 0: return p0;  case 1: return p1;  case 2: return p2;
    case 3: return p3;  case 4: return p4;  case 5: return p5;
    default: return p6;
  }
}

// ---------------------------------------------------------------- kernels

__global__ void f32_to_f16_kernel(const float* __restrict__ s,
                                  _Float16* __restrict__ d, int n) {
  int i = blockIdx.x * blockDim.x + threadIdx.x;
  if (i < n) d[i] = (_Float16)s[i];
}

// e = sigmoid(x_weather @ w_e^T + b_e) -> f16. grid=(kH/64, kN/16), block=128
__global__ void __launch_bounds__(128)
gemm_e_kernel(const _Float16* __restrict__ wth16,
              const _Float16* __restrict__ we16,
              const float* __restrict__ b_e,
              _Float16* __restrict__ e16) {
  const int lane = threadIdx.x & 31;
  const int wave = threadIdx.x >> 5;
  const int h0 = (blockIdx.x * 4 + wave) * 16;
  const int n0 = blockIdx.y * 16;

  v8f acc{};
  #pragma unroll
  for (int k0 = 0; k0 < kW; k0 += 32) {
    v16h a = load_frag(wth16, kW, n0, k0, lane);
    v16h b = load_frag(we16, kW, h0, k0, lane);
    acc = wmma_f16(a, b, acc);
  }
  const int nn = lane & 15;
  const int ms = (lane >> 4) << 3;
  const int h  = h0 + nn;
  const float be = b_e[h];
  #pragma unroll
  for (int r = 0; r < 8; ++r) {
    size_t idx = (size_t)(n0 + ms + r) * kH + h;
    e16[idx] = (_Float16)sigmoid_fast(acc[r] + be);
  }
}

// Five fused GEMMs: pre_ho, pre_i, pre_g, pre_f, pre_o.
// Block = 128 (4 waves). All waves share h0 = blockIdx.x*16; wave w handles
// M-subtile n0 = (blockIdx.y*4 + w)*16. Weight (B) tiles are staged once per
// block into LDS (async-copy path) and reused by all 4 waves.
// grid = (kH/16, kN/64) = (64, 135)
__global__ void __launch_bounds__(128)
gemm_main_kernel(const _Float16* __restrict__ x16,
                 const _Float16* __restrict__ e16,
                 const _Float16* __restrict__ wix, const _Float16* __restrict__ wgx,
                 const _Float16* __restrict__ wfx, const _Float16* __restrict__ wox,
                 const _Float16* __restrict__ wie, const _Float16* __restrict__ wfe,
                 const _Float16* __restrict__ woe,
                 const _Float16* __restrict__ wd,  const _Float16* __restrict__ wwk,
                 const _Float16* __restrict__ wm,
                 const float* __restrict__ b_i, const float* __restrict__ b_g,
                 const float* __restrict__ b_f, const float* __restrict__ b_o,
                 float* __restrict__ ph, float* __restrict__ pi,
                 float* __restrict__ pg, float* __restrict__ pf,
                 float* __restrict__ po) {
  __shared__ __align__(16) char sB[7 * kTileB];
  const int tid  = threadIdx.x;
  const int lane = tid & 31;
  const int wave = tid >> 5;
  const int h0 = blockIdx.x * 16;
  const int n0 = (blockIdx.y * 4 + wave) * 16;
  const int day = n0 / kT;   // uniform per wave (96 % 16 == 0)

  v8f ai{}, ag{}, af{}, ao{}, aho{};

  // ---- K over I: 7 staged weight tiles per k-step
  for (int k0 = 0; k0 < kI; k0 += 32) {
    __syncthreads();   // previous tiles fully consumed
    for (int c = tid; c < 7 * 64; c += 128) {
      const int t   = c >> 6;
      const int r   = (c >> 2) & 15;
      const int col = c & 3;
      const _Float16* g = pick7(wix, wgx, wfx, wox, wd, wwk, wm, t) +
                          (size_t)(h0 + r) * kI + k0 + col * 8;
      stage_chunk(g, sB + t * kTileB + r * kTileRowB + col * 16);
    }
#if USE_ASYNC_LDS
    __builtin_amdgcn_s_wait_asynccnt(0);
#endif
    __syncthreads();   // staged tiles visible

    v16h a = load_frag(x16, kI, n0, k0, lane);
    ai = wmma_f16(a, frag_from_lds(sB, 0, lane), ai);
    ag = wmma_f16(a, frag_from_lds(sB, 1, lane), ag);
    af = wmma_f16(a, frag_from_lds(sB, 2, lane), af);
    ao = wmma_f16(a, frag_from_lds(sB, 3, lane), ao);
    if (day >= 1) {   // h_d = x[d-1]
      aho = wmma_f16(load_frag(x16, kI, n0 - 1 * kT, k0, lane),
                     frag_from_lds(sB, 4, lane), aho);
    }
    if (day >= 7) {   // h_w = x[d-6]
      aho = wmma_f16(load_frag(x16, kI, n0 - 6 * kT, k0, lane),
                     frag_from_lds(sB, 5, lane), aho);
    }
    if (day >= 28) {  // h_m = x[(d-29) % 90]  (d==28 wraps to day 89)
      const int r29 = (day >= 29) ? (n0 - 29 * kT) : (n0 - 29 * kT + kN);
      aho = wmma_f16(load_frag(x16, kI, r29, k0, lane),
                     frag_from_lds(sB, 6, lane), aho);
    }
  }

  // ---- K over H: e-gate contributions, 3 staged weight tiles per k-step
  for (int k0 = 0; k0 < kH; k0 += 32) {
    __syncthreads();
    for (int c = tid; c < 3 * 64; c += 128) {
      const int t   = c >> 6;
      const int r   = (c >> 2) & 15;
      const int col = c & 3;
      const _Float16* g = pick7(wie, wfe, woe, woe, woe, woe, woe, t) +
                          (size_t)(h0 + r) * kH + k0 + col * 8;
      stage_chunk(g, sB + t * kTileB + r * kTileRowB + col * 16);
    }
#if USE_ASYNC_LDS
    __builtin_amdgcn_s_wait_asynccnt(0);
#endif
    __syncthreads();

    v16h ae = load_frag(e16, kH, n0, k0, lane);
    ai = wmma_f16(ae, frag_from_lds(sB, 0, lane), ai);
    af = wmma_f16(ae, frag_from_lds(sB, 1, lane), af);
    ao = wmma_f16(ae, frag_from_lds(sB, 2, lane), ao);
  }

  // ---- Epilogue: C layout -> lane = (m>=8)*16 + n, vgpr r = m & 7
  const int nn = lane & 15;
  const int ms = (lane >> 4) << 3;
  const int h  = h0 + nn;
  const float bi = b_i[h], bg = b_g[h], bf = b_f[h], bo = b_o[h];
  #pragma unroll
  for (int r = 0; r < 8; ++r) {
    size_t idx = (size_t)(n0 + ms + r) * kH + h;
    ph[idx] = aho[r];
    pi[idx] = ai[r] + bi;
    pg[idx] = ag[r] + bg;
    pf[idx] = af[r] + bf;
    po[idx] = ao[r] + bo;
  }
}

// Sequential scan: 1 workgroup, 1024 threads (32 wave32 waves).
// h, c, h_o live in LDS; recurrence weights stream from L2 (20 MB << 192 MB).
__global__ void __launch_bounds__(1024)
scan_kernel(const float* __restrict__ ph, const float* __restrict__ pi,
            const float* __restrict__ pg, const float* __restrict__ pf,
            const float* __restrict__ po,
            const float* __restrict__ w_t,  const float* __restrict__ w_ih,
            const float* __restrict__ w_gh, const float* __restrict__ w_fo,
            const float* __restrict__ w_oh,
            float* __restrict__ hs, float* __restrict__ cs) {
  __shared__ float sh[kH];
  __shared__ float sc[kH];
  __shared__ float sho[kH];
  const int tid = threadIdx.x;

  sh[tid] = 0.0f;
  sc[tid] = 0.0f;
  __syncthreads();

  const float4* wt_r  = (const float4*)(w_t  + (size_t)tid * kH);
  const float4* woh_r = (const float4*)(w_oh + (size_t)tid * kH);
  const float4* wih_r = (const float4*)(w_ih + (size_t)tid * kH);
  const float4* wgh_r = (const float4*)(w_gh + (size_t)tid * kH);
  const float4* wfo_r = (const float4*)(w_fo + (size_t)tid * kH);

  for (int n = 0; n < kN; ++n) {
    const size_t base = (size_t)n * kH + tid;
    // prefetch next step's pre-activation rows into cache
    if (n + 1 < kN) {
      __builtin_prefetch(ph + base + kH, 0, 0);
      __builtin_prefetch(pi + base + kH, 0, 0);
      __builtin_prefetch(pg + base + kH, 0, 0);
      __builtin_prefetch(pf + base + kH, 0, 0);
      __builtin_prefetch(po + base + kH, 0, 0);
    }

    // stage 1: w_t@h and w_oh@h
    float at = 0.0f, aoh = 0.0f;
    const float4* hv = (const float4*)sh;
    #pragma unroll 4
    for (int k = 0; k < kH / 4; ++k) {
      float4 w1 = wt_r[k];
      float4 w2 = woh_r[k];
      float4 h4 = hv[k];
      at  += w1.x * h4.x + w1.y * h4.y + w1.z * h4.z + w1.w * h4.w;
      aoh += w2.x * h4.x + w2.y * h4.y + w2.z * h4.z + w2.w * h4.w;
    }
    const float hoval = sigmoid_fast(ph[base] + 2.0f * at);  // ref adds w_t@h twice
    const float oval  = sigmoid_fast(po[base] + aoh);
    const float cprev = sc[tid];
    sho[tid] = hoval;
    __syncthreads();

    // stage 2: w_ih@h_o, w_gh@h_o, w_fo@h_o
    float aiv = 0.0f, agv = 0.0f, afv = 0.0f;
    const float4* hov = (const float4*)sho;
    #pragma unroll 4
    for (int k = 0; k < kH / 4; ++k) {
      float4 w1 = wih_r[k];
      float4 w2 = wgh_r[k];
      float4 w3 = wfo_r[k];
      float4 h4 = hov[k];
      aiv += w1.x * h4.x + w1.y * h4.y + w1.z * h4.z + w1.w * h4.w;
      agv += w2.x * h4.x + w2.y * h4.y + w2.z * h4.z + w2.w * h4.w;
      afv += w3.x * h4.x + w3.y * h4.y + w3.z * h4.z + w3.w * h4.w;
    }
    const float iv = sigmoid_fast(pi[base] + aiv);
    const float gv = tanh_fast(pg[base] + agv);
    const float fv = sigmoid_fast(pf[base] + afv);
    const float cn = fv * cprev + iv * gv;
    const float hn = oval * tanh_fast(cn);
    hs[base] = hn;
    cs[base] = cn;
    __syncthreads();
    sh[tid] = hn;
    sc[tid] = cn;
    __syncthreads();
  }
}

// ---------------------------------------------------------------- launch

extern "C" void kernel_launch(void* const* d_in, const int* in_sizes, int n_in,
                              void* d_out, int out_size, void* d_ws, size_t ws_size,
                              hipStream_t stream) {
  const float* x_input   = (const float*)d_in[0];
  const float* x_weather = (const float*)d_in[1];
  const float* w_ix = (const float*)d_in[2];
  const float* w_ih = (const float*)d_in[3];
  const float* w_ie = (const float*)d_in[4];
  const float* b_i  = (const float*)d_in[5];
  const float* w_fx = (const float*)d_in[6];
  const float* w_fo = (const float*)d_in[7];
  const float* w_fe = (const float*)d_in[8];
  const float* b_f  = (const float*)d_in[9];
  const float* w_ox = (const float*)d_in[10];
  const float* w_oh = (const float*)d_in[11];
  const float* w_oe = (const float*)d_in[12];
  const float* b_o  = (const float*)d_in[13];
  const float* w_gx = (const float*)d_in[14];
  const float* w_gh = (const float*)d_in[15];
  const float* b_g  = (const float*)d_in[16];
  const float* w_d  = (const float*)d_in[17];
  const float* w_w  = (const float*)d_in[18];
  const float* w_m  = (const float*)d_in[19];
  const float* w_t  = (const float*)d_in[20];
  const float* w_e  = (const float*)d_in[21];
  const float* b_e  = (const float*)d_in[22];

  float* hs = (float*)d_out;
  float* cs = hs + (size_t)kN * kH;

  size_t off = 0;
  auto carve = [&](size_t bytes) -> void* {
    void* p = (char*)d_ws + off;
    off += (bytes + 255) & ~(size_t)255;
    return p;
  };
  _Float16* x16   = (_Float16*)carve((size_t)kN * kI * 2);
  _Float16* wth16 = (_Float16*)carve((size_t)kN * kW * 2);
  _Float16* e16   = (_Float16*)carve((size_t)kN * kH * 2);
  _Float16* wix16 = (_Float16*)carve((size_t)kH * kI * 2);
  _Float16* wgx16 = (_Float16*)carve((size_t)kH * kI * 2);
  _Float16* wfx16 = (_Float16*)carve((size_t)kH * kI * 2);
  _Float16* wox16 = (_Float16*)carve((size_t)kH * kI * 2);
  _Float16* wd16  = (_Float16*)carve((size_t)kH * kI * 2);
  _Float16* ww16  = (_Float16*)carve((size_t)kH * kI * 2);
  _Float16* wm16  = (_Float16*)carve((size_t)kH * kI * 2);
  _Float16* wie16 = (_Float16*)carve((size_t)kH * kH * 2);
  _Float16* wfe16 = (_Float16*)carve((size_t)kH * kH * 2);
  _Float16* woe16 = (_Float16*)carve((size_t)kH * kH * 2);
  _Float16* we16  = (_Float16*)carve((size_t)kH * kW * 2);
  float* ph = (float*)carve((size_t)kN * kH * 4);
  float* pi = (float*)carve((size_t)kN * kH * 4);
  float* pg = (float*)carve((size_t)kN * kH * 4);
  float* pf = (float*)carve((size_t)kN * kH * 4);
  float* po = (float*)carve((size_t)kN * kH * 4);

  auto conv = [&](const float* s, _Float16* t, size_t n) {
    f32_to_f16_kernel<<<dim3((unsigned)((n + 255) / 256)), 256, 0, stream>>>(s, t, (int)n);
  };
  conv(x_input,   x16,   (size_t)kN * kI);
  conv(x_weather, wth16, (size_t)kN * kW);
  conv(w_ix, wix16, (size_t)kH * kI);
  conv(w_gx, wgx16, (size_t)kH * kI);
  conv(w_fx, wfx16, (size_t)kH * kI);
  conv(w_ox, wox16, (size_t)kH * kI);
  conv(w_d,  wd16,  (size_t)kH * kI);
  conv(w_w,  ww16,  (size_t)kH * kI);
  conv(w_m,  wm16,  (size_t)kH * kI);
  conv(w_ie, wie16, (size_t)kH * kH);
  conv(w_fe, wfe16, (size_t)kH * kH);
  conv(w_oe, woe16, (size_t)kH * kH);
  conv(w_e,  we16,  (size_t)kH * kW);

  gemm_e_kernel<<<dim3(kH / 64, kN / 16), 128, 0, stream>>>(wth16, we16, b_e, e16);

  gemm_main_kernel<<<dim3(kH / 16, kN / 64), 128, 0, stream>>>(
      x16, e16,
      wix16, wgx16, wfx16, wox16,
      wie16, wfe16, woe16,
      wd16, ww16, wm16,
      b_i, b_g, b_f, b_o,
      ph, pi, pg, pf, po);

  scan_kernel<<<1, 1024, 0, stream>>>(ph, pi, pg, pf, po,
                                      w_t, w_ih, w_gh, w_fo, w_oh,
                                      hs, cs);
}
